// ISA_34565896798370
// MI455X (gfx1250) — matrix operand order repaired and checked by reading, not compiled, not run on hardware
//
#include <hip/hip_runtime.h>
#include <hip/hip_bf16.h>

// ---------------- problem constants ----------------
#define B_   4
#define N_   2048
#define IN_  384
#define S_   8
#define D_   192
#define G_   64          // D/3
#define SIGMA_ 5.0f
#define SCALE_ 0.07216878364870322f   // 192^-0.5

typedef __attribute__((ext_vector_type(16))) __bf16 v16bf;
typedef __attribute__((ext_vector_type(8)))  float  v8f;

__device__ __forceinline__ __bf16 f32_to_bf16(float f) {
  union { float f; unsigned int u; } v; v.f = f;
  unsigned int r = v.u + 0x7FFFu + ((v.u >> 16) & 1u);   // round-to-nearest-even
  union { unsigned short u; __bf16 h; } o; o.u = (unsigned short)(r >> 16);
  return o.h;
}

// =====================================================================
// WMMA GEMM:  Y[M x Nc] = act(A[M x K] @ W[Nc x K]^T + bias)
// M % 32 == 0, Nc % 192 == 0, K % 32 == 0 (true for all call sites:
// Nc in {192,384,576,768}).  Block: 256 threads = 8 waves;
// tile = 32 rows x 192 cols, so A is read only Nc/192 times total.
// Each wave: 1 row-half x 3 col-tiles -> 3 WMMAs per K-stage off one
// A fragment.
// =====================================================================
__global__ void __launch_bounds__(256)
gemm_bf16_wmma(const float* __restrict__ A, const float* __restrict__ W,
               const float* __restrict__ bias, float* __restrict__ Y,
               int M, int Nc, int K, int act)
{
  __shared__ __bf16 la[32][34];    // A tile  32 x 32  (+pad)
  __shared__ __bf16 lw[192][34];   // W tile 192 x 32  (+pad)

  const int tid  = threadIdx.x;
  const int lane = tid & 31;
  const int wave = tid >> 5;
  const int rw   = wave & 1;            // row half  (0/1)
  const int cw   = wave >> 1;           // col group (0..3), 48 cols each
  const int rowBase = blockIdx.y * 32;
  const int colBase = blockIdx.x * 192;

  // per-thread staging source pointers (advance by 32 each K-step)
  const int aIdx = tid * 4;                 // 1024 = 32x32, 4/thread
  const int aR = aIdx >> 5, aC = aIdx & 31;
  const float* srcA = A + (size_t)(rowBase + aR) * K + aC;

  v8f acc0 = {}, acc1 = {}, acc2 = {};

  for (int k0 = 0; k0 < K; k0 += 32) {
    // ---- stage A tile (coalesced f32, convert to bf16) ----
    {
      const float* p = srcA + k0;
      #pragma unroll
      for (int i = 0; i < 4; ++i) la[aR][aC + i] = f32_to_bf16(p[i]);
      if (k0 + 32 < K) __builtin_prefetch(p + 32, 0, 3);   // global_prefetch_b8
    }
    // ---- stage W tile: 6144 elems = 192x32, 24/thread in 6 passes ----
    #pragma unroll
    for (int it = 0; it < 6; ++it) {
      int idx = (tid + it * 256) * 4;
      int r = idx >> 5, c = idx & 31;
      const float* src = W + (size_t)(colBase + r) * K + k0 + c;
      #pragma unroll
      for (int i = 0; i < 4; ++i) lw[r][c + i] = f32_to_bf16(src[i]);
    }
    __syncthreads();

    // A fragment (16x32): lanes 0-15 -> K 0-15,
    // lanes 16-31 -> K 16-31; row M = lane & 15.
    v16bf a;
    {
      const int m   = 16 * rw + (lane & 15);
      const int kbA = (lane < 16) ? 0 : 16;
      #pragma unroll
      for (int i = 0; i < 8; ++i) {
        a[i]     = la[m][kbA + i];
        a[i + 8] = la[m][kbA + 8 + i];
      }
    }
    // B fragments (32x16): column N = lane & 15 within tile; lanes 0-15
    // hold K 0-15, lanes 16-31 hold K 16-31.  B[k][n] == W[n][k].
    const int kbB = (lane < 16) ? 0 : 16;
    const int nb  = 48 * cw + (lane & 15);
    v16bf b0, b1, b2;
    #pragma unroll
    for (int i = 0; i < 16; ++i) {
      b0[i] = lw[nb][kbB + i];
      b1[i] = lw[nb + 16][kbB + i];
      b2[i] = lw[nb + 32][kbB + i];
    }
    acc0 = __builtin_amdgcn_wmma_f32_16x16x32_bf16(false, a, false, b0,
                                                   (short)0, acc0, false, false);
    acc1 = __builtin_amdgcn_wmma_f32_16x16x32_bf16(false, a, false, b1,
                                                   (short)0, acc1, false, false);
    acc2 = __builtin_amdgcn_wmma_f32_16x16x32_bf16(false, a, false, b2,
                                                   (short)0, acc2, false, false);
    __syncthreads();
  }

  // C/D layout: VGPR r -> M = r (lanes 0-15) or r+8 (lanes 16-31); N = lane&15
  const int mo = rowBase + 16 * rw + ((lane < 16) ? 0 : 8);
  #pragma unroll
  for (int t = 0; t < 3; ++t) {
    const v8f acc = (t == 0) ? acc0 : (t == 1) ? acc1 : acc2;
    const int n = colBase + 48 * cw + 16 * t + (lane & 15);
    const float bv = bias ? bias[n] : 0.0f;
    #pragma unroll
    for (int r = 0; r < 8; ++r) {
      float v = acc[r] + bv;
      if (act == 1) v = fmaxf(v, 0.0f);
      Y[(size_t)(mo + r) * Nc + n] = v;
    }
  }
}

// =====================================================================
// LayerNorm, wave per row (wave32).  flags&1 -> nan_to_num(0, +1, -1)
// =====================================================================
__global__ void ln_kernel(const float* __restrict__ x, const float* __restrict__ g,
                          const float* __restrict__ b, float* __restrict__ y,
                          int rows, int cols, int flags)
{
  int row  = blockIdx.x * (blockDim.x >> 5) + (threadIdx.x >> 5);
  int lane = threadIdx.x & 31;
  if (row >= rows) return;
  const float* xr = x + (size_t)row * cols;
  float s = 0.f, ss = 0.f;
  for (int c = lane; c < cols; c += 32) { float v = xr[c]; s += v; ss += v * v; }
  #pragma unroll
  for (int o = 16; o > 0; o >>= 1) {
    s  += __shfl_xor(s,  o, 32);
    ss += __shfl_xor(ss, o, 32);
  }
  float mean = s / (float)cols;
  float var  = ss / (float)cols - mean * mean;
  float inv  = rsqrtf(var + 1e-5f);
  float* yr = y + (size_t)row * cols;
  for (int c = lane; c < cols; c += 32) {
    float v = (xr[c] - mean) * inv * g[c] + b[c];
    if (flags & 1) {
      if (isnan(v)) v = 0.f;
      else if (isinf(v)) v = (v > 0.f) ? 1.f : -1.f;
    }
    yr[c] = v;
  }
}

// =====================================================================
// Geo positional encoding fused with Xk = kx + pos, Xv = vx + pos.
// Thread per (b,s,j,g).
// =====================================================================
__global__ void pos_add_kernel(const float* __restrict__ pc,
                               const float* __restrict__ Sp,
                               const float* __restrict__ Ssb,
                               const float* __restrict__ basis,
                               const float* __restrict__ kx,
                               const float* __restrict__ vx,
                               float* __restrict__ Xk, float* __restrict__ Xv)
{
  size_t idx = (size_t)blockIdx.x * blockDim.x + threadIdx.x;
  const size_t total = (size_t)B_ * S_ * N_ * G_;
  if (idx >= total) return;
  int g = (int)(idx % G_);
  size_t bsj = idx / G_;
  int j = (int)(bsj % N_);
  size_t bs = bsj / N_;
  int b = (int)(bs / S_);

  float inv_freq = powf(100.0f, (float)g * (1.0f / 96.0f)); // GEO_BASE^(2g/D)
  float th[3]; float th2 = 0.f;
  #pragma unroll
  for (int c = 0; c < 3; ++c) {
    float ssv = Ssb[bs * 3 + c];
    ssv = fminf(fmaxf(ssv, 0.3f), 5.0f) + 0.1f;
    float r = (pc[((size_t)b * N_ + j) * 3 + c] - Sp[bs * 3 + c]) /
              (ssv * SIGMA_ + 1e-4f);
    r = fminf(fmaxf(r, -3.f), 3.f);
    th[c] = r * inv_freq;
    th2 += th[c] * th[c];
  }
  float Theta = sqrtf(th2) * (1.0f / 3.0f);
  float cs = __cosf(0.5f * Theta);
  float sf = __sinf(0.5f * Theta) / (3.0f * Theta + 1e-6f);
  float u0 = sf * th[0], u1 = sf * th[1], u2 = sf * th[2];
  float v0 = basis[g * 3 + 0], v1 = basis[g * 3 + 1], v2 = basis[g * 3 + 2];
  float uv0 = u1 * v2 - u2 * v1;
  float uv1 = u2 * v0 - u0 * v2;
  float uv2 = u0 * v1 - u1 * v0;
  float uuv0 = u1 * uv2 - u2 * uv1;
  float uuv1 = u2 * uv0 - u0 * uv2;
  float uuv2 = u0 * uv1 - u1 * uv0;
  float rot[3];
  rot[0] = v0 + 2.f * cs * uv0 + 2.f * uuv0;
  rot[1] = v1 + 2.f * cs * uv1 + 2.f * uuv1;
  rot[2] = v2 + 2.f * cs * uv2 + 2.f * uuv2;

  size_t base  = (bs * N_ + j) * D_ + (size_t)g * 3;
  size_t tbase = ((size_t)b * N_ + j) * D_ + (size_t)g * 3;
  #pragma unroll
  for (int c = 0; c < 3; ++c) {
    Xk[base + c] = kx[tbase + c] + rot[c];
    Xv[base + c] = vx[tbase + c] + rot[c];
  }
}

// ================= dots: wave per (b,s,j) =================
__global__ void dots_kernel(const float* __restrict__ q,
                            const float* __restrict__ k,
                            float* __restrict__ dots)
{
  size_t w = (size_t)blockIdx.x * (blockDim.x >> 5) + (threadIdx.x >> 5);
  int lane = threadIdx.x & 31;
  if (w >= (size_t)B_ * S_ * N_) return;
  size_t bs = w / N_;
  const float* kr = k + w * D_;
  const float* qr = q + bs * D_;
  float s = 0.f;
  #pragma unroll
  for (int d = 0; d < D_ / 32; ++d) s += qr[lane + d * 32] * kr[lane + d * 32];
  #pragma unroll
  for (int o = 16; o > 0; o >>= 1) s += __shfl_xor(s, o, 32);
  if (lane == 0)
    dots[w] = fminf(fmaxf(s * SCALE_, -30.f), 30.f);
}

// ======== softmax over slots (axis=1) + 1e-8, thread per (b,j) ========
__global__ void softmax_slots_kernel(float* __restrict__ dots)
{
  int idx = blockIdx.x * blockDim.x + threadIdx.x;   // b*N + j
  if (idx >= B_ * N_) return;
  int b = idx / N_, j = idx % N_;
  float v[S_]; float mx = -1e30f;
  #pragma unroll
  for (int s = 0; s < S_; ++s) {
    v[s] = dots[((size_t)b * S_ + s) * N_ + j];
    mx = fmaxf(mx, v[s]);
  }
  float sum = 0.f;
  #pragma unroll
  for (int s = 0; s < S_; ++s) { v[s] = __expf(v[s] - mx); sum += v[s]; }
  float inv = 1.0f / sum;
  #pragma unroll
  for (int s = 0; s < S_; ++s)
    dots[((size_t)b * S_ + s) * N_ + j] = v[s] * inv + 1e-8f;
}

// ======== renormalize over j: block per (b,s) ========
__global__ void attn_norm_kernel(float* __restrict__ attn)
{
  __shared__ float red[256];
  size_t base = (size_t)blockIdx.x * N_;
  float s = 0.f;
  for (int j = threadIdx.x; j < N_; j += 256) s += attn[base + j];
  red[threadIdx.x] = s;
  __syncthreads();
  for (int o = 128; o > 0; o >>= 1) {
    if ((int)threadIdx.x < o) red[threadIdx.x] += red[threadIdx.x + o];
    __syncthreads();
  }
  float inv = 1.0f / (red[0] + 1e-8f);
  for (int j = threadIdx.x; j < N_; j += 256) attn[base + j] *= inv;
}

// ==== updates[b,s,d] = sum_j v*attn ; S_p = sum_j pc*attn. block/(b,s) ====
__global__ void updates_kernel(const float* __restrict__ v,
                               const float* __restrict__ attn,
                               const float* __restrict__ pc,
                               float* __restrict__ upd, float* __restrict__ Sp)
{
  int bs = blockIdx.x;                 // 0..31
  int b  = bs / S_;
  int d  = threadIdx.x;                // 0..191
  const float* ar = attn + (size_t)bs * N_;
  const float* vr = v + (size_t)bs * N_ * D_;
  float u = 0.f, sp = 0.f;
  for (int j = 0; j < N_; ++j) {
    float a = ar[j];
    u += vr[(size_t)j * D_ + d] * a;
    if (d < 3) sp += pc[((size_t)b * N_ + j) * 3 + d] * a;
  }
  upd[(size_t)bs * D_ + d] = u;
  if (d < 3) Sp[bs * 3 + d] = sp;
}

// ==== S_s = sqrt(sum_j (pc-S_p)^2 * attn + 1e-6). one block, 96 threads ====
__global__ void ss_kernel(const float* __restrict__ pc,
                          const float* __restrict__ Sp,
                          const float* __restrict__ attn,
                          float* __restrict__ Ssb)
{
  int t = threadIdx.x;
  if (t >= B_ * S_ * 3) return;
  int c = t % 3, bs = t / 3, b = bs / S_;
  float sp = Sp[bs * 3 + c];
  const float* ar = attn + (size_t)bs * N_;
  float acc = 0.f;
  for (int j = 0; j < N_; ++j) {
    float d = pc[((size_t)b * N_ + j) * 3 + c] - sp;
    acc += d * d * ar[j];
  }
  Ssb[bs * 3 + c] = sqrtf(acc + 1e-6f);
}

// ==== GRU combine (in-place on sl) ====
__global__ void gru_combine_kernel(const float* __restrict__ gi,
                                   const float* __restrict__ gh,
                                   float* __restrict__ sl)
{
  int idx = blockIdx.x * blockDim.x + threadIdx.x;   // B*S*D
  if (idx >= B_ * S_ * D_) return;
  int d = idx % D_; int bs = idx / D_;
  const float* gir = gi + (size_t)bs * 3 * D_;
  const float* ghr = gh + (size_t)bs * 3 * D_;
  float r = 1.f / (1.f + __expf(-(gir[d] + ghr[d])));
  float z = 1.f / (1.f + __expf(-(gir[D_ + d] + ghr[D_ + d])));
  float n = tanhf(gir[2 * D_ + d] + r * ghr[2 * D_ + d]);
  float h = sl[idx];
  sl[idx] = (1.f - z) * n + z * h;
}

// ==== misc small kernels ====
__global__ void bcast_slots_kernel(const float* __restrict__ slots, float* __restrict__ sl)
{
  int idx = blockIdx.x * blockDim.x + threadIdx.x;   // B*S*D
  if (idx >= B_ * S_ * D_) return;
  sl[idx] = slots[idx % (S_ * D_)];
}
__global__ void init_stats_kernel(const float* __restrict__ Sp0,
                                  const float* __restrict__ Ss0,
                                  float* __restrict__ Sp, float* __restrict__ Ssb)
{
  int idx = threadIdx.x;                             // B*S*3 = 96
  if (idx >= B_ * S_ * 3) return;
  Sp[idx]  = Sp0[idx % (S_ * 3)];
  Ssb[idx] = Ss0[idx % (S_ * 3)];
}
__global__ void add_kernel(float* __restrict__ y, const float* __restrict__ x, int n)
{
  int idx = blockIdx.x * blockDim.x + threadIdx.x;
  if (idx < n) y[idx] += x[idx];
}

// =====================================================================
extern "C" void kernel_launch(void* const* d_in, const int* in_sizes, int n_in,
                              void* d_out, int out_size, void* d_ws, size_t ws_size,
                              hipStream_t stream)
{
  (void)in_sizes; (void)n_in; (void)out_size; (void)ws_size;
  const float* inputs   = (const float*)d_in[0];
  const float* pcoords  = (const float*)d_in[1];
  const float* slots    = (const float*)d_in[2];
  const float* Ss_init  = (const float*)d_in[3];
  const float* Sp_init  = (const float*)d_in[4];
  const float* basis    = (const float*)d_in[5];
  const float* Q_w      = (const float*)d_in[6];
  const float* K_w      = (const float*)d_in[7];
  const float* V_w      = (const float*)d_in[8];
  const float* norm_g   = (const float*)d_in[9];
  const float* norm_b   = (const float*)d_in[10];
  const float* gru_wih  = (const float*)d_in[11];
  const float* gru_whh  = (const float*)d_in[12];
  const float* gru_bih  = (const float*)d_in[13];
  const float* gru_bhh  = (const float*)d_in[14];
  const float* mlp_ng   = (const float*)d_in[15];
  const float* mlp_nb   = (const float*)d_in[16];
  const float* mlp_w1   = (const float*)d_in[17];
  const float* mlp_b1   = (const float*)d_in[18];
  const float* mlp_w2   = (const float*)d_in[19];
  const float* mlp_b2   = (const float*)d_in[20];
  const float* f_w1     = (const float*)d_in[21];
  const float* f_b1     = (const float*)d_in[22];
  const float* f_w2     = (const float*)d_in[23];
  const float* f_b2     = (const float*)d_in[24];
  const float* im_ln1_g = (const float*)d_in[25];
  const float* im_ln1_b = (const float*)d_in[26];
  const float* im_w1    = (const float*)d_in[27];
  const float* im_b1    = (const float*)d_in[28];
  const float* im_w2    = (const float*)d_in[29];
  const float* im_b2    = (const float*)d_in[30];
  const float* im_ln2_g = (const float*)d_in[31];
  const float* im_ln2_b = (const float*)d_in[32];
  const float* final_w  = (const float*)d_in[33];
  const float* final_b  = (const float*)d_in[34];
  float* out = (float*)d_out;

  // ---- workspace layout ----
  float* ws = (float*)d_ws;
  size_t off = 0;
  auto alloc = [&](size_t n) { float* p = ws + off; off += (n + 63) & ~(size_t)63; return p; };
  const size_t BN = (size_t)B_ * N_;
  const size_t BSND = (size_t)B_ * S_ * N_ * D_;
  float* xln  = alloc(BN * IN_);     // LN(inputs), also reused as pre-LN temp
  float* him  = alloc(BN * IN_);     // im hidden
  float* temb = alloc(BN * D_);      // token_emb
  float* kx   = alloc(BN * D_);
  float* vx   = alloc(BN * D_);
  float* Xk   = alloc(BSND);
  float* Xv   = alloc(BSND);
  float* Hb   = alloc(BSND);
  float* Kb   = alloc(BSND);
  float* Vb   = alloc(BSND);
  float* attn = alloc((size_t)B_ * S_ * N_);
  float* sl   = alloc(B_ * S_ * D_);
  float* sn   = alloc(B_ * S_ * D_);
  float* qb   = alloc(B_ * S_ * D_);
  float* upd  = alloc(B_ * S_ * D_);
  float* Sp   = alloc(B_ * S_ * 3);
  float* Ssb  = alloc(B_ * S_ * 3);
  float* gi   = alloc((size_t)B_ * S_ * 3 * D_);
  float* gh   = alloc((size_t)B_ * S_ * 3 * D_);
  float* mn   = alloc(B_ * S_ * D_);
  float* mlph = alloc((size_t)B_ * S_ * 4 * D_);
  float* mlpo = alloc(B_ * S_ * D_);

  auto gemm = [&](const float* A, const float* W, const float* bias, float* Y,
                  int M, int Nc, int K, int act) {
    dim3 grid(Nc / 192, M / 32);
    gemm_bf16_wmma<<<grid, 256, 0, stream>>>(A, W, bias, Y, M, Nc, K, act);
  };

  // ---- token embedding ----
  ln_kernel<<<(B_ * N_) / 4, 128, 0, stream>>>(inputs, im_ln1_g, im_ln1_b, xln,
                                               B_ * N_, IN_, 0);
  gemm(xln, im_w1, im_b1, him, B_ * N_, IN_, IN_, 1);          // relu
  gemm(him, im_w2, im_b2, xln, B_ * N_, D_, IN_, 0);           // reuse xln
  ln_kernel<<<(B_ * N_) / 4, 128, 0, stream>>>(xln, im_ln2_g, im_ln2_b, temb,
                                               B_ * N_, D_, 1); // nan_to_num
  gemm(temb, K_w, nullptr, kx, B_ * N_, D_, D_, 0);
  gemm(temb, V_w, nullptr, vx, B_ * N_, D_, D_, 0);

  // ---- init slot state ----
  bcast_slots_kernel<<<(B_ * S_ * D_ + 255) / 256, 256, 0, stream>>>(slots, sl);
  init_stats_kernel<<<1, 128, 0, stream>>>(Sp_init, Ss_init, Sp, Ssb);

  // ---- slot-attention iterations (t = 3 provably dead for output: skipped) ----
  const size_t posThreads = (size_t)B_ * S_ * N_ * G_;
  for (int t = 0; t < 3; ++t) {
    ln_kernel<<<(B_ * S_) / 4, 128, 0, stream>>>(sl, norm_g, norm_b, sn,
                                                 B_ * S_, D_, 0);
    gemm(sn, Q_w, nullptr, qb, B_ * S_, D_, D_, 0);

    pos_add_kernel<<<(unsigned)((posThreads + 255) / 256), 256, 0, stream>>>(
        pcoords, Sp, Ssb, basis, kx, vx, Xk, Xv);

    // k = f_net(kx + pos), v = f_net(vx + pos)
    gemm(Xk, f_w1, f_b1, Hb, (int)(B_ * S_ * N_), D_, D_, 1);
    gemm(Hb, f_w2, f_b2, Kb, (int)(B_ * S_ * N_), D_, D_, 0);
    gemm(Xv, f_w1, f_b1, Hb, (int)(B_ * S_ * N_), D_, D_, 1);
    gemm(Hb, f_w2, f_b2, Vb, (int)(B_ * S_ * N_), D_, D_, 0);

    dots_kernel<<<(B_ * S_ * N_) / 8, 256, 0, stream>>>(qb, Kb, attn);
    softmax_slots_kernel<<<(B_ * N_) / 256, 256, 0, stream>>>(attn);
    attn_norm_kernel<<<B_ * S_, 256, 0, stream>>>(attn);

    updates_kernel<<<B_ * S_, D_, 0, stream>>>(Vb, attn, pcoords, upd, Sp);
    ss_kernel<<<1, 128, 0, stream>>>(pcoords, Sp, attn, Ssb);

    // GRU + residual MLP
    gemm(upd, gru_wih, gru_bih, gi, B_ * S_, 3 * D_, D_, 0);
    gemm(sl,  gru_whh, gru_bhh, gh, B_ * S_, 3 * D_, D_, 0);
    gru_combine_kernel<<<(B_ * S_ * D_ + 255) / 256, 256, 0, stream>>>(gi, gh, sl);

    ln_kernel<<<(B_ * S_) / 4, 128, 0, stream>>>(sl, mlp_ng, mlp_nb, mn,
                                                 B_ * S_, D_, 0);
    gemm(mn, mlp_w1, mlp_b1, mlph, B_ * S_, 4 * D_, D_, 1);
    gemm(mlph, mlp_w2, mlp_b2, mlpo, B_ * S_, D_, 4 * D_, 0);
    add_kernel<<<(B_ * S_ * D_ + 255) / 256, 256, 0, stream>>>(sl, mlpo, B_ * S_ * D_);
  }

  // ---- output projection straight into d_out ----
  gemm(sl, final_w, final_b, out, B_ * S_, D_, D_, 0);
}
